// BertSelfAttention_57157424775555
// MI455X (gfx1250) — matrix operand order repaired and checked by reading, not compiled, not run on hardware
//
#include <hip/hip_runtime.h>
#include <hip/hip_bf16.h>

#define BB   16
#define LL   512
#define HIDN 768
#define HH   12
#define DD   64
#define NS   5
#define BH   (BB*HH)

typedef __attribute__((ext_vector_type(16))) __bf16 v16bf;
typedef __attribute__((ext_vector_type(8)))  __bf16 v8bf;
typedef __attribute__((ext_vector_type(4)))  __bf16 v4bf;
typedef __attribute__((ext_vector_type(2)))  __bf16 v2bf;
typedef __attribute__((ext_vector_type(8)))  float  v8f;
typedef __attribute__((ext_vector_type(4)))  float  f32x4;
typedef unsigned int u32x4 __attribute__((ext_vector_type(4)));
typedef int          i32x4 __attribute__((ext_vector_type(4)));
typedef int          i32x8 __attribute__((ext_vector_type(8)));

__device__ __forceinline__ v8f wmma_bf16(v16bf a, v16bf b, v8f c) {
  return __builtin_amdgcn_wmma_f32_16x16x32_bf16(false, a, false, b, (short)0, c, false, false);
}

// A-fragment: 16x32 bf16 from row-major [M][K] (ld = K stride) -> two 16B loads.
__device__ __forceinline__ v16bf load_a(const __bf16* base, int ld, int lane) {
  const __bf16* p = base + (lane & 15) * ld + ((lane >> 4) << 3);
  v8bf lo = *(const v8bf*)p;
  v8bf hi = *(const v8bf*)(p + 16);
  return __builtin_shufflevector(lo, hi, 0,1,2,3,4,5,6,7,8,9,10,11,12,13,14,15);
}

// B-fragment: B[k][n] = M[n][k], M row-major [N][K] -> two 16B loads.
__device__ __forceinline__ v16bf load_bT(const __bf16* base, int ld, int lane) {
  const __bf16* p = base + (lane & 15) * ld + ((lane >> 4) << 3);
  v8bf lo = *(const v8bf*)p;
  v8bf hi = *(const v8bf*)(p + 16);
  return __builtin_shufflevector(lo, hi, 0,1,2,3,4,5,6,7,8,9,10,11,12,13,14,15);
}

__device__ __forceinline__ v4bf pack4(f32x4 f) {
  v4bf r;
  r[0] = (__bf16)f.x; r[1] = (__bf16)f.y; r[2] = (__bf16)f.z; r[3] = (__bf16)f.w;
  return r;
}

// ---- Tensor Data Mover: contiguous 1-D bf16 copy, global -> LDS ----
typedef __attribute__((address_space(3))) char lds_char;
__device__ __forceinline__ unsigned lds_off(void* p) {
  return (unsigned)(unsigned long long)(lds_char*)p;
}

__device__ __forceinline__ void tdm_load_contig_bf16(unsigned lds_addr, const void* gptr,
                                                     unsigned nelems /* 2B elems, <65536 */) {
  unsigned long long ga = (unsigned long long)gptr;
  u32x4 g0;
  g0[0] = 1u;                                                // count=1 valid user descriptor
  g0[1] = lds_addr;                                          // lds_addr [63:32]
  g0[2] = (unsigned)(ga & 0xFFFFFFFFu);                      // global_addr lo
  g0[3] = (unsigned)((ga >> 32) & 0x01FFFFFFu) | (2u << 30); // global_addr hi, type=2
  i32x8 g1;
  g1[0] = 0x00010000;                                        // data_size=1 (2B)
  g1[1] = (int)((nelems & 0xFFFFu) << 16);                   // tensor_dim0[15:0] @ [63:48]
  g1[2] = (int)(((nelems >> 16) & 0xFFFFu)) | (1 << 16);     // tensor_dim0[31:16], tensor_dim1=1
  g1[3] = (int)((nelems & 0xFFFFu) << 16);                   // tile_dim0 @ [127:112]
  g1[4] = 0;                                                 // tile_dim1/2 = 0 (1-D)
  g1[5] = (int)nelems;                                       // tensor_dim0_stride lo
  g1[6] = 0;
  g1[7] = 0;
  i32x4 z4 = {0, 0, 0, 0};
  i32x8 z8 = {0, 0, 0, 0, 0, 0, 0, 0};
  __builtin_amdgcn_tensor_load_to_lds(g0, g1, z4, z4, z8, 0);
}

// ---------------- Kernel 1: QKV projection ----------------
__global__ __launch_bounds__(256) void qkv_proj_kernel(
    const float* __restrict__ hs,
    const float* __restrict__ Wq, const float* __restrict__ bq,
    const float* __restrict__ Wk, const float* __restrict__ bk,
    const float* __restrict__ Wv, const float* __restrict__ bv,
    __bf16* __restrict__ qh, __bf16* __restrict__ kh, __bf16* __restrict__ vh)
{
  __shared__ __bf16 aT[64][32];   // [m][k]
  __shared__ __bf16 bT[64][32];   // [n][k]
  const int m0 = blockIdx.x * 64;
  const int n0 = blockIdx.y * 64;
  const int z  = blockIdx.z;
  const float* W    = (z == 0) ? Wq : (z == 1) ? Wk : Wv;
  const float* bias = (z == 0) ? bq : (z == 1) ? bk : bv;
  __bf16* dst       = (z == 0) ? qh : (z == 1) ? kh : vh;

  const int tid = threadIdx.x, lane = tid & 31, wave = tid >> 5;
  const int tr = wave >> 1, hc = wave & 1;
  // vectorized staging coords: 512 float4 items per tile, 2 per thread
  const int sr = tid >> 3, sk = (tid & 7) * 4;   // rows 0..31 handled at +0, +32

  v8f acc0 = {}, acc1 = {};
  for (int kc = 0; kc < HIDN; kc += 32) {
    __syncthreads();
#pragma unroll
    for (int half = 0; half < 2; ++half) {
      int r = sr + half * 32;
      f32x4 ha = *(const f32x4*)&hs[(m0 + r) * HIDN + kc + sk];
      f32x4 wb = *(const f32x4*)&W [(n0 + r) * HIDN + kc + sk];
      *(v4bf*)&aT[r][sk] = pack4(ha);
      *(v4bf*)&bT[r][sk] = pack4(wb);
    }
    __syncthreads();
    v16bf af = load_a(&aT[tr*16][0], 32, lane);
    v16bf b0 = load_bT(&bT[hc*32][0],      32, lane);
    v16bf b1 = load_bT(&bT[hc*32 + 16][0], 32, lane);
    acc0 = wmma_bf16(af, b0, acc0);
    acc1 = wmma_bf16(af, b1, acc1);
  }
#pragma unroll
  for (int t = 0; t < 2; ++t) {
    v8f acc = t ? acc1 : acc0;
    int tc = hc*2 + t;
#pragma unroll
    for (int r = 0; r < 8; ++r) {
      int mm = r + ((lane >> 4) << 3);
      int gm = m0 + tr*16 + mm;
      int gn = n0 + tc*16 + (lane & 15);
      float val = acc[r] + bias[gn];
      int bb = gm >> 9, ll = gm & (LL - 1);
      int hh = gn >> 6, dd = gn & (DD - 1);
      dst[(((long)bb*HH + hh)*LL + ll)*DD + dd] = (__bf16)val;
    }
  }
}

// ---------------- Kernel 2: QB_s = q @ bili[s,h] ----------------
__global__ __launch_bounds__(256) void qb_proj_kernel(
    const __bf16* __restrict__ qh, const float* __restrict__ bili,
    __bf16* __restrict__ qb)
{
  __shared__ __bf16 aT[64][64];   // [m][p]
  __shared__ __bf16 bTq[64][64];  // [q][p]
  const int m0 = blockIdx.x * 64;
  const int bh = blockIdx.y;
  const int s  = blockIdx.z;
  const int h  = bh % HH;
  const int tid = threadIdx.x, lane = tid & 31, wave = tid >> 5;
  const int tr = wave >> 1, hc = wave & 1;

  // q tile: 16B copies (512 items, 2 per thread)
  for (int i = tid; i < 64*8; i += 256) {
    int r = i >> 3, c0 = (i & 7) * 8;
    *(v8bf*)&aT[r][c0] = *(const v8bf*)&qh[((long)bh*LL + m0 + r)*DD + c0];
  }
  // bili: float4 loads, transposed bf16 scatter (1024 items, 4 per thread)
  for (int i = tid; i < 64*16; i += 256) {
    int p = i >> 4, q0 = (i & 15) * 4;
    f32x4 f = *(const f32x4*)&bili[(((long)s*HH + h)*DD + p)*DD + q0];
    bTq[q0 + 0][p] = (__bf16)f.x;
    bTq[q0 + 1][p] = (__bf16)f.y;
    bTq[q0 + 2][p] = (__bf16)f.z;
    bTq[q0 + 3][p] = (__bf16)f.w;
  }
  __syncthreads();
  v8f acc0 = {}, acc1 = {};
#pragma unroll
  for (int c = 0; c < 2; ++c) {
    v16bf af = load_a(&aT[tr*16][c*32], 64, lane);
    v16bf b0 = load_bT(&bTq[hc*32][c*32],      64, lane);
    v16bf b1 = load_bT(&bTq[hc*32 + 16][c*32], 64, lane);
    acc0 = wmma_bf16(af, b0, acc0);
    acc1 = wmma_bf16(af, b1, acc1);
  }
#pragma unroll
  for (int t = 0; t < 2; ++t) {
    v8f acc = t ? acc1 : acc0;
    int tc = hc*2 + t;
#pragma unroll
    for (int r = 0; r < 8; ++r) {
      int mm = r + ((lane >> 4) << 3);
      qb[((((long)s*BH + bh)*LL) + m0 + tr*16 + mm)*DD + tc*16 + (lane & 15)] = (__bf16)acc[r];
    }
  }
}

// ---------------- Kernel 3: fused structured attention ----------------
__global__ __launch_bounds__(256) void attn_kernel(
    const __bf16* __restrict__ qh, const __bf16* __restrict__ kh,
    const __bf16* __restrict__ vh, const __bf16* __restrict__ qb,
    const float* __restrict__ smask, const float* __restrict__ amask,
    const float* __restrict__ absb, float* __restrict__ out)
{
  __shared__ __bf16 kTile[64][64];   // [key][d]  <- TDM
  __shared__ __bf16 vStage[64][64];  // [key][d]  <- TDM
  __shared__ __bf16 vT[64][64];      // [d][key]
  __shared__ float  sS[64][64];
  __shared__ __bf16 pT[64][64];      // [row][key]
  __shared__ float  rowM[64], rowL[64], rowA[64];

  const int m0 = blockIdx.x * 64;
  const int bh = blockIdx.y;
  const int b  = bh / HH, h = bh % HH;
  const int tid = threadIdx.x, lane = tid & 31, wave = tid >> 5;
  const int tr = wave >> 1, hc = wave & 1;

  // Register-cache A-operands: q + 5 biaffine QB_s
  v16bf aQ[2], aQB[NS][2];
  const __bf16* qbase = qh + ((long)bh*LL + m0 + tr*16) * DD;
#pragma unroll
  for (int c = 0; c < 2; ++c) aQ[c] = load_a(qbase + c*32, DD, lane);
#pragma unroll
  for (int s = 0; s < NS; ++s) {
    const __bf16* qbb = qb + (((long)s*BH + bh)*LL + m0 + tr*16) * DD;
#pragma unroll
    for (int c = 0; c < 2; ++c) aQB[s][c] = load_a(qbb + c*32, DD, lane);
  }
  float absv[NS];
#pragma unroll
  for (int s = 0; s < NS; ++s) absv[s] = absb[s*HH + h];

  if (tid < 64) { rowM[tid] = -1e30f; rowL[tid] = 0.f; }
  v8f accO0 = {}, accO1 = {};

  for (int j = 0; j < 8; ++j) {
    const int k0 = j * 64;
    __syncthreads();                       // previous tile fully consumed
    if (wave == 0) {
      tdm_load_contig_bf16(lds_off(&kTile[0][0]),  kh + ((long)bh*LL + k0)*DD, 64*64);
      tdm_load_contig_bf16(lds_off(&vStage[0][0]), vh + ((long)bh*LL + k0)*DD, 64*64);
      __builtin_amdgcn_s_wait_tensorcnt(0);
    }
    __syncthreads();                       // TDM data visible to all waves
    // transpose V in LDS: vT[d][key]
    for (int i = tid; i < 64*8; i += 256) {
      int n = i & 63, d0 = (i >> 6) << 3;
      v8bf vv = *(const v8bf*)&vStage[n][d0];
#pragma unroll
      for (int e = 0; e < 8; ++e) vT[d0 + e][n] = vv[e];
    }
    __syncthreads();

    // S = q@K^T + sum_s mask_s * (QB_s@K^T + abs_bias_s)
#pragma unroll
    for (int t = 0; t < 2; ++t) {
      int tc = hc*2 + t;
      v16bf bk0 = load_bT(&kTile[tc*16][0],  64, lane);
      v16bf bk1 = load_bT(&kTile[tc*16][32], 64, lane);
      v8f sacc = {};
      sacc = wmma_bf16(aQ[0], bk0, sacc);
      sacc = wmma_bf16(aQ[1], bk1, sacc);
      float comb[8];
#pragma unroll
      for (int r = 0; r < 8; ++r) comb[r] = sacc[r];
#pragma unroll
      for (int s = 0; s < NS; ++s) {
        v8f bacc = {};
        bacc = wmma_bf16(aQB[s][0], bk0, bacc);
        bacc = wmma_bf16(aQB[s][1], bk1, bacc);
#pragma unroll
        for (int r = 0; r < 8; ++r) {
          int mm   = r + ((lane >> 4) << 3);
          int grow = m0 + tr*16 + mm;
          int gcol = k0 + tc*16 + (lane & 15);
          float mv = smask[(((long)s*BB + b)*LL + grow)*LL + gcol];
          comb[r] += (bacc[r] + absv[s]) * mv;
        }
      }
#pragma unroll
      for (int r = 0; r < 8; ++r) {
        int mm = r + ((lane >> 4) << 3);
        int lr = tr*16 + mm;
        int lc = tc*16 + (lane & 15);
        sS[lr][lc] = comb[r] * 0.125f + amask[b*LL + k0 + lc];   // 1/sqrt(64)
      }
    }
    __syncthreads();

    // online softmax over this 64-key tile
    if (tid < 64) {
      float mold = rowM[tid];
      float mnew = mold;
      for (int c = 0; c < 64; ++c) mnew = fmaxf(mnew, sS[tid][c]);
      float alpha = __expf(mold - mnew);
      float sum = 0.f;
      for (int c = 0; c < 64; c += 2) {
        float p0 = __expf(sS[tid][c]     - mnew);
        float p1 = __expf(sS[tid][c + 1] - mnew);
        v2bf pk; pk[0] = (__bf16)p0; pk[1] = (__bf16)p1;
        *(v2bf*)&pT[tid][c] = pk;
        sum += p0 + p1;
      }
      rowL[tid] = rowL[tid] * alpha + sum;
      rowM[tid] = mnew;
      rowA[tid] = alpha;
    }
    __syncthreads();

    // O = O*alpha + P@V
#pragma unroll
    for (int r = 0; r < 8; ++r) {
      int mm = r + ((lane >> 4) << 3);
      float al = rowA[tr*16 + mm];
      accO0[r] *= al;
      accO1[r] *= al;
    }
    v16bf p0 = load_a(&pT[tr*16][0],  64, lane);
    v16bf p1 = load_a(&pT[tr*16][32], 64, lane);
#pragma unroll
    for (int t = 0; t < 2; ++t) {
      int tc = hc*2 + t;
      v16bf bv0 = load_bT(&vT[tc*16][0],  64, lane);
      v16bf bv1 = load_bT(&vT[tc*16][32], 64, lane);
      v8f& ao = t ? accO1 : accO0;
      ao = wmma_bf16(p0, bv0, ao);
      ao = wmma_bf16(p1, bv1, ao);
    }
  }
  __syncthreads();

  // normalize, write out[b, l, h*D + d] (fp32)
#pragma unroll
  for (int t = 0; t < 2; ++t) {
    int tc = hc*2 + t;
    v8f ao = t ? accO1 : accO0;
#pragma unroll
    for (int r = 0; r < 8; ++r) {
      int mm   = r + ((lane >> 4) << 3);
      int lrow = m0 + tr*16 + mm;
      float invl = 1.0f / rowL[tr*16 + mm];
      int d = tc*16 + (lane & 15);
      out[(((long)b*LL + lrow)*HIDN) + h*DD + d] = ao[r] * invl;
    }
  }
}

extern "C" void kernel_launch(void* const* d_in, const int* in_sizes, int n_in,
                              void* d_out, int out_size, void* d_ws, size_t ws_size,
                              hipStream_t stream) {
  const float* hs    = (const float*)d_in[0];
  const float* amask = (const float*)d_in[1];
  const float* smask = (const float*)d_in[2];
  const float* Wq    = (const float*)d_in[3];
  const float* bq    = (const float*)d_in[4];
  const float* Wk    = (const float*)d_in[5];
  const float* bk    = (const float*)d_in[6];
  const float* Wv    = (const float*)d_in[7];
  const float* bv    = (const float*)d_in[8];
  const float* bili  = (const float*)d_in[9];
  const float* absb  = (const float*)d_in[10];
  float* out = (float*)d_out;

  char* ws = (char*)d_ws;
  const size_t perQKV = (size_t)BH * LL * DD * sizeof(__bf16);  // 12.58 MB
  __bf16* qh = (__bf16*)(ws);
  __bf16* kh = (__bf16*)(ws + perQKV);
  __bf16* vh = (__bf16*)(ws + 2 * perQKV);
  __bf16* qb = (__bf16*)(ws + 3 * perQKV);                       // 5x perQKV

  qkv_proj_kernel<<<dim3((BB*LL)/64, HIDN/64, 3), 256, 0, stream>>>(
      hs, Wq, bq, Wk, bk, Wv, bv, qh, kh, vh);
  qb_proj_kernel<<<dim3(LL/64, BH, NS), 256, 0, stream>>>(qh, bili, qb);
  attn_kernel<<<dim3(LL/64, BH), 256, 0, stream>>>(
      qh, kh, vh, qb, smask, amask, absb, out);
}
